// ConvolutionalSelfAttention_1486058684519
// MI455X (gfx1250) — matrix operand order repaired and checked by reading, not compiled, not run on hardware
//
#include <hip/hip_runtime.h>

// CDNA5 / gfx1250 implementation of ConvolutionalSelfAttention.
//
// Strategy (see analysis): per (batch, window-slab) block, stage the whole
// 16x16x64 image in LDS as L2-normalized f16 vectors (32KB), then compute
// sim = X_all(256x64) . X_loc(64x9)^T per window with v_wmma_f32_16x16x32_f16
// (16 M-tiles x 2 K-steps = 32 WMMAs/window), fusing the masked softmax
// statistics (sum e, sum s*e) straight off the f32 accumulator tiles.
// Final out[n,c] = sum_k W[k]*x_k[c] uses original fp32 data from global.

typedef __attribute__((ext_vector_type(16))) _Float16 v16h;
typedef __attribute__((ext_vector_type(8)))  float    v8f;
typedef __attribute__((ext_vector_type(8)))  unsigned int v8u;

union ABfrag { v8u u; v16h h; };

static constexpr int kH    = 16;
static constexpr int kW    = 16;
static constexpr int kC    = 64;
static constexpr int kB    = 32;
static constexpr int kCH   = 14;          // H - K + 1
static constexpr int kNWin = kCH * kCH;   // 196 windows per image
static constexpr int kNPos = kH * kW;     // 256 positions
static constexpr float kTemp = 10.0f;

__device__ __forceinline__ unsigned int pack_f16x2(float a, float b) {
  _Float16 ha = (_Float16)a;
  _Float16 hb = (_Float16)b;
  unsigned short ua = __builtin_bit_cast(unsigned short, ha);
  unsigned short ub = __builtin_bit_cast(unsigned short, hb);
  return (unsigned int)ua | ((unsigned int)ub << 16);
}

__global__ __launch_bounds__(256) void csa_wmma_kernel(
    const float* __restrict__ X,    // (B, 256, 64) fp32
    const float* __restrict__ Wg,   // (64,) fp32
    const float* __restrict__ bg,   // (1,) fp32
    float* __restrict__ out)        // (B, 196, 64) fp32
{
  // Normalized image, f16 packed 2-per-u32: sXn[p*32 + c/2]
  __shared__ unsigned int sXn[kNPos * (kC / 2)];
  __shared__ float sS[kNPos];   // s_p = x_p . Wg + bg
  __shared__ float sWg[kC];

  const int b   = blockIdx.x;
  const int tid = threadIdx.x;

  if (tid < kC) sWg[tid] = Wg[tid];
  __syncthreads();

  // ---------------- Phase 1: normalize + gate scalar, one thread per position
  {
    const int p = tid;
    const float* xp = X + ((size_t)b * kNPos + p) * kC;
    float ss = 0.f, dot = 0.f;
    for (int c = 0; c < kC; c += 4) {
      float4 v = *(const float4*)(xp + c);
      ss  += v.x * v.x + v.y * v.y + v.z * v.z + v.w * v.w;
      dot += v.x * sWg[c] + v.y * sWg[c + 1] + v.z * sWg[c + 2] + v.w * sWg[c + 3];
    }
    const float inv = 1.0f / fmaxf(sqrtf(ss), 1e-12f);
    sS[p] = dot + bg[0];
    unsigned int* dstp = &sXn[p * 32];
    for (int c = 0; c < kC; c += 2) {
      float2 v = *(const float2*)(xp + c);
      dstp[c >> 1] = pack_f16x2(v.x * inv, v.y * inv);
    }
  }
  __syncthreads();

  // ---------------- Phase 2: per-window WMMA + fused masked softmax
  const int wave = tid >> 5;
  const int lane = tid & 31;
  const int col  = lane & 15;   // N index (local position k, valid < 9)
  const int hh   = lane >> 4;   // K half-group selector per ISA layout

  const int nBeg = blockIdx.y * 28;   // 7 slabs * 28 windows = 196
  const int nEnd = nBeg + 28;

  for (int n = nBeg + wave; n < nEnd; n += 8) {   // uniform per wave
    const int wi = n / kCH;
    const int wj = n % kCH;

    // B fragments (32x16 f16, K x N): lane col holds its column's channels.
    // K = 16*hh + 2v, 2v+1 in u32 element v  -> u32 idx (8*hh + v), +16 for K>=32.
    ABfrag b0, b1;
    if (col < 9) {
      const int q = (wi + col / 3) * kW + (wj + col % 3);
      const unsigned int* qp = &sXn[q * 32];
      #pragma unroll
      for (int v = 0; v < 8; ++v) {
        b0.u[v] = qp[8 * hh + v];
        b1.u[v] = qp[16 + 8 * hh + v];
      }
    } else {
      #pragma unroll
      for (int v = 0; v < 8; ++v) { b0.u[v] = 0u; b1.u[v] = 0u; }
    }

    float se = 0.f;   // sum of exp over valid g (this lane's column)
    float ws = 0.f;   // sum of s_g * exp

    for (int t = 0; t < 16; ++t) {    // 16 M-tiles cover all 256 positions
      // A fragment (16x32 f16): lane (16h + m) holds row M=m.
      // VGPR v: K pair (16h+2v) (contiguous K order, matching B).
      const unsigned int* pp = &sXn[(16 * t + col) * 32];
      ABfrag a0, a1;
      #pragma unroll
      for (int v = 0; v < 4; ++v) {
        a0.u[v]     = pp[8 * hh + v];
        a0.u[v + 4] = pp[4 + 8 * hh + v];
        a1.u[v]     = pp[16 + 8 * hh + v];
        a1.u[v + 4] = pp[20 + 8 * hh + v];
      }

      v8f acc = {};
      acc = __builtin_amdgcn_wmma_f32_16x16x32_f16(
          false, a0.h, false, b0.h, (short)0, acc, false, false);
      acc = __builtin_amdgcn_wmma_f32_16x16x32_f16(
          false, a1.h, false, b1.h, (short)0, acc, false, false);

      // D layout: lane (16h + nIdx) VGPR v = D[M = 8h+v][N = nIdx]
      #pragma unroll
      for (int v = 0; v < 8; ++v) {
        const int p2 = 16 * t + 8 * hh + v;
        const unsigned dr = (unsigned)(p2 >> 4) - (unsigned)wi;
        const unsigned dc = (unsigned)(p2 & 15) - (unsigned)wj;
        const bool isLocal = (dr < 3u) && (dc < 3u);   // p2 in LOC(n): masked
        const float e = isLocal ? 0.f : __expf(kTemp * acc[v]);
        se += e;
        ws += sS[p2] * e;
      }
    }

    // Combine the two K-half lanes holding the same column.
    se += __shfl_xor(se, 16, 32);
    ws += __shfl_xor(ws, 16, 32);
    const float Wfin = ws / se;   // W[k] on lanes with col == k (k < 9)

    // out[n, c] = sum_k W[k] * x_orig[p_k, c]; each lane does 2 channels.
    float2 accp; accp.x = 0.f; accp.y = 0.f;
    #pragma unroll
    for (int k = 0; k < 9; ++k) {
      const float Wk = __shfl(Wfin, k, 32);
      const int pk = (wi + k / 3) * kW + (wj + k % 3);
      const float2 xv =
          *(const float2*)(X + ((size_t)b * kNPos + pk) * kC + 2 * lane);
      accp.x += Wk * xv.x;
      accp.y += Wk * xv.y;
    }
    *(float2*)(out + ((size_t)b * kNWin + n) * kC + 2 * lane) = accp;
  }
}

extern "C" void kernel_launch(void* const* d_in, const int* in_sizes, int n_in,
                              void* d_out, int out_size, void* d_ws, size_t ws_size,
                              hipStream_t stream) {
  const float* X  = (const float*)d_in[0];   // (32,16,16,64) fp32
  const float* Wg = (const float*)d_in[1];   // (64,1) fp32
  const float* bg = (const float*)d_in[2];   // (1,) fp32
  float* out = (float*)d_out;                // (32,14,14,64) fp32

  dim3 grid(kB, 7);   // 7 slabs of 28 windows each = 196
  csa_wmma_kernel<<<grid, 256, 0, stream>>>(X, Wg, bg, out);
}
